// QuantONet_64201171141238
// MI455X (gfx1250) — 8-bit matrix operand order repaired and checked by reading, compiled, not run on hardware
//
#include <hip/hip_runtime.h>
#include <stdint.h>


#define BATCH 4096

typedef __attribute__((ext_vector_type(8))) int v8i_t;
union V8 { v8i_t v; int e[8]; unsigned u[8]; };

enum { F_S0 = 0, F_SW1 = 1, F_SW2, F_SW3, F_SW4, F_SWF, F_SWB, F_SWC, F_SWL,
       F_SA1 = 10, F_SA2, F_SA3, F_SA4, F_SA5,
       F_SC1 = 16, F_SC2, F_SC3, F_SC4, F_SCF, F_SCB, F_SCC, F_SCL,
       F_SQB = 24, F_SQC, F_SQL,
       F_R0 = 27, F_RA1 = 28, F_RA2, F_RA3, F_RA4, F_RA5,
       F_RQB = 33, F_RQC, F_RQL };
enum { I_ABSX = 0, I_MAX1, I_MAX2, I_MAX3, I_MAX4, I_MAX5, I_ABSB, I_ABSC, I_ABSL,
       BI1 = 32, BI2 = 64, BI3 = 128, BI4 = 192, BIF = 320, BIB = 576, BIC = 580, BIL = 582 };

__global__ void init_k(unsigned* p) {
    p[(size_t)blockIdx.x * 256 + threadIdx.x] = 0u;
}

__global__ void absmax_k(const float* __restrict__ x, long n, unsigned* bits) {
    __shared__ float red[256];
    float m = 0.f;
    for (long i = (long)blockIdx.x * 256 + threadIdx.x; i < n; i += (long)gridDim.x * 256)
        m = fmaxf(m, fabsf(x[i]));
    red[threadIdx.x] = m; __syncthreads();
    for (int off = 128; off; off >>= 1) {
        if ((int)threadIdx.x < off) red[threadIdx.x] = fmaxf(red[threadIdx.x], red[threadIdx.x + off]);
        __syncthreads();
    }
    if (threadIdx.x == 0) atomicMax(bits, __float_as_uint(red[0]));
}

__global__ void scale_div_k(const unsigned* bits, float* s, float* rs, float d) {
    float v = fmaxf(__uint_as_float(*bits) / d, 1e-8f);
    *s = v; *rs = 1.0f / v;
}

__global__ void scale_act_k(const float* sc, const int* imax, float* s, float* rs) {
    int m = *imax; if (m < 0) m = 0;
    float v = fmaxf((*sc) * (float)m / 15.f, 1e-8f);
    *s = v; *rs = 1.0f / v;
}

__global__ void bias_prep_k(const float* __restrict__ b, const float* s_in,
                            const float* sw, float* sc_out, int* bi, int n) {
    float sc = (*s_in) * (*sw);
    if (threadIdx.x == 0) *sc_out = sc;
    if ((int)threadIdx.x < n) bi[threadIdx.x] = (int)rintf(b[threadIdx.x] / sc);
}

__global__ void quantx_k(const float* __restrict__ x, const float* rs,
                         int8_t* __restrict__ q, long n) {
    long i = (long)blockIdx.x * 256 + threadIdx.x;
    if (i >= n) return;
    int ic = (int)(i & 3); long t = i >> 2;
    int iw = (int)(t % 48); t /= 48;
    int ih = (int)(t % 48); int b = (int)(t / 48);
    int8_t qv = 0;
    if (ic < 3) {
        float v = x[(((long)b * 3 + ic) * 48 + ih) * 48 + iw];
        float r = rintf(v * (*rs));
        r = fminf(7.f, fmaxf(-8.f, r));
        qv = (int8_t)(int)r;
    }
    q[i] = qv;
}

__global__ void pack_plain_k(const float* __restrict__ w, int total,
                             float* sw_out, int8_t* __restrict__ q8) {
    __shared__ float red[256]; __shared__ float ssw;
    float m = 0.f;
    for (int i = threadIdx.x; i < total; i += 256) m = fmaxf(m, fabsf(w[i]));
    red[threadIdx.x] = m; __syncthreads();
    for (int off = 128; off; off >>= 1) {
        if ((int)threadIdx.x < off) red[threadIdx.x] = fmaxf(red[threadIdx.x], red[threadIdx.x + off]);
        __syncthreads();
    }
    if (threadIdx.x == 0) { ssw = fmaxf(red[0] / 7.f, 1e-8f); *sw_out = ssw; }
    __syncthreads();
    float s = ssw;
    for (int i = threadIdx.x; i < total; i += 256) {
        float t = rintf(w[i] / s);
        t = fminf(7.f, fmaxf(-8.f, t));
        q8[i] = (int8_t)(int)t;
    }
}

__global__ void pack_frag_k(const float* __restrict__ w, int OC, int K,
                            int IC, int ICreal, int KH, int KW, int flip,
                            float* sw_out, int* __restrict__ frag) {
    __shared__ float red[256]; __shared__ float ssw;
    int Ksrc = ICreal * KH * KW;
    int total = OC * Ksrc;
    float m = 0.f;
    for (int i = threadIdx.x; i < total; i += 256) m = fmaxf(m, fabsf(w[i]));
    red[threadIdx.x] = m; __syncthreads();
    for (int off = 128; off; off >>= 1) {
        if ((int)threadIdx.x < off) red[threadIdx.x] = fmaxf(red[threadIdx.x], red[threadIdx.x + off]);
        __syncthreads();
    }
    if (threadIdx.x == 0) { ssw = fmaxf(red[0] / 7.f, 1e-8f); *sw_out = ssw; }
    __syncthreads();
    float s = ssw;
    int ksteps = (K + 63) >> 6, ntiles = OC >> 4;
    int ndw = ntiles * ksteps * 32 * 8;
    for (int d = threadIdx.x; d < ndw; d += 256) {
        int v = d & 7, r = d >> 3;
        int lane = r & 31; r >>= 5;
        int ks = r % ksteps, nt = r / ksteps;
        int half = lane >> 4, n = nt * 16 + (lane & 15);
        unsigned dw = 0;
        for (int bb = 0; bb < 4; bb++) {
            int k = ks * 64 + ((v >> 2) << 5) + (half << 4) + ((v & 3) << 2) + bb;
            int q = 0;
            if (k < K) {
                int ic = k % IC, sidx = k / IC;
                if (ic < ICreal) {
                    int i = sidx / KW, j = sidx % KW;
                    int src = flip ? (ic * KH * KW + j * KH + i)
                                   : (ic * KH * KW + i * KW + j);
                    float t = rintf(w[n * Ksrc + src] / s);
                    t = fminf(7.f, fmaxf(-8.f, t));
                    q = (int)t;
                }
            }
            dw |= (unsigned)(q & 0xFF) << (bb * 8);
        }
        frag[d] = (int)dw;
    }
}

__global__ void pool_k(const int8_t* __restrict__ in, int8_t* __restrict__ out,
                       long total, int C, int IH, int IW, int OH, int OW, int kk, int st) {
    long i = (long)blockIdx.x * 256 + threadIdx.x;
    if (i >= total) return;
    int c = (int)(i % C); long t = i / C;
    int ow = (int)(t % OW); t /= OW;
    int oh = (int)(t % OH); long b = t / OH;
    int best = 0;
    for (int a = 0; a < kk; a++) {
        int ih = oh * st + a; if (ih >= IH) break;
        for (int bb = 0; bb < kk; bb++) {
            int iw = ow * st + bb; if (iw >= IW) continue;
            int v = in[((b * IH + ih) * IW + iw) * C + c];
            if (v > best) best = v;
        }
    }
    out[i] = (int8_t)best;
}

template <bool SIGNED_A, bool IS_FC, int K, int OC, int IC, int LG_IC,
          int IH, int IW, int OH, int OW, int KH, int KW>
__global__ void __launch_bounds__(128)
gemmq_k(const int8_t* __restrict__ act, const int* __restrict__ wfrag,
        const int* __restrict__ bias_i, const float* sc_p, int* maxacc,
        const float* rs_p, int8_t* __restrict__ out, int M, int mode) {
    constexpr int KSTEPS = (K + 63) >> 6;
    __shared__ int ldsb[2 * KSTEPS * 256];

    const int lane  = threadIdx.x & 31;
    const int wave  = threadIdx.x >> 5;
    const int mtile = blockIdx.x * 4 + wave;
    const int nt0   = blockIdx.y * 2;
    const int half  = lane >> 4;
    const int mrow  = mtile * 16 + (lane & 15);
    const int ncol0 = nt0 * 16 + (lane & 15);

    {
        const int4* src = (const int4*)(wfrag + (size_t)nt0 * KSTEPS * 256);
        int4* dst = (int4*)ldsb;
        for (int i = threadIdx.x; i < 2 * KSTEPS * 64; i += 128) dst[i] = src[i];
    }
    __syncthreads();

    const int8_t* abase;
    if (IS_FC) {
        abase = act + (size_t)mrow * K;
    } else {
        constexpr int PO = OH * OW;
        int b0 = mrow / PO; int r = mrow % PO;
        int oh0 = r / OW, ow0 = r % OW;
        abase = act + ((size_t)(b0 * IH + oh0) * IW + ow0) * IC;
    }

    V8 acc0, acc1;
#pragma unroll
    for (int i = 0; i < 8; i++) { acc0.e[i] = 0; acc1.e[i] = 0; }

#pragma unroll
    for (int ks = 0; ks < KSTEPS; ks++) {
        V8 a;
#pragma unroll
        for (int v = 0; v < 8; v++) {
            int kc = ks * 64 + ((v >> 2) << 5) + (half << 4) + ((v & 3) << 2);
            bool valid = kc < K;
            int off;
            if (IS_FC) {
                off = kc;
            } else {
                int ic = kc & (IC - 1);
                int t  = kc >> LG_IC;
                int kh = t / KW, kw = t % KW;
                off = (kh * IW + kw) * IC + ic;
            }
            off = valid ? off : 0;
            unsigned dw = *(const unsigned*)(abase + off);
            a.u[v] = valid ? dw : 0u;
        }
        V8 b0f, b1f;
        {
            const int4* l0 = (const int4*)&ldsb[((0 * KSTEPS + ks) * 32 + lane) * 8];
            const int4* l1 = (const int4*)&ldsb[((1 * KSTEPS + ks) * 32 + lane) * 8];
            int4 x0 = l0[0], x1 = l0[1], y0 = l1[0], y1 = l1[1];
            b0f.e[0] = x0.x; b0f.e[1] = x0.y; b0f.e[2] = x0.z; b0f.e[3] = x0.w;
            b0f.e[4] = x1.x; b0f.e[5] = x1.y; b0f.e[6] = x1.z; b0f.e[7] = x1.w;
            b1f.e[0] = y0.x; b1f.e[1] = y0.y; b1f.e[2] = y0.z; b1f.e[3] = y0.w;
            b1f.e[4] = y1.x; b1f.e[5] = y1.y; b1f.e[6] = y1.z; b1f.e[7] = y1.w;
        }
        acc0.v = __builtin_amdgcn_wmma_i32_16x16x64_iu8(SIGNED_A, a.v, true, b0f.v, acc0.v, false, false);
        acc1.v = __builtin_amdgcn_wmma_i32_16x16x64_iu8(SIGNED_A, a.v, true, b1f.v, acc1.v, false, false);
    }

    const int bia0 = bias_i[ncol0];
    const int bia1 = bias_i[ncol0 + 16];
    if (mode == 0) {
        int lm = acc0.e[0] + bia0;
#pragma unroll
        for (int v = 1; v < 8; v++) { int t = acc0.e[v] + bia0; if (t > lm) lm = t; }
#pragma unroll
        for (int v = 0; v < 8; v++) { int t = acc1.e[v] + bia1; if (t > lm) lm = t; }
        for (int off = 16; off > 0; off >>= 1) {
            int o = __shfl_xor(lm, off);
            if (o > lm) lm = o;
        }
        if (lane == 0) atomicMax(maxacc, lm);
    } else {
        float sc = *sc_p;
        float rs = *rs_p;
        float mul = sc * rs;
#pragma unroll
        for (int v = 0; v < 8; v++) {
            int m = mtile * 16 + v + half * 8;
            size_t obase;
            if (IS_FC) {
                obase = (size_t)m * OC;
            } else {
                constexpr int PO = OH * OW;
                int b = m / PO; int r = m % PO;
                int oh = r / OW, ow = r % OW;
                obase = ((size_t)(b * OH + oh) * OW + ow) * OC;
            }
            float t0 = fmaxf((float)(acc0.e[v] + bia0) * mul, 0.f);
            int q0 = (int)rintf(t0);
            q0 = q0 > 15 ? 15 : q0;
            out[obase + ncol0] = (int8_t)q0;
            float t1 = fmaxf((float)(acc1.e[v] + bia1) * mul, 0.f);
            int q1 = (int)rintf(t1);
            q1 = q1 > 15 ? 15 : q1;
            out[obase + ncol0 + 16] = (int8_t)q1;
        }
    }
}

__global__ void head_k(const int8_t* __restrict__ a5, const int8_t* __restrict__ qw,
                       const int* __restrict__ bi, const float* sc_p, unsigned* maxbits,
                       const float* sq_p, const float* rq_p,
                       float* __restrict__ out, int N, int mode) {
    long idx = (long)blockIdx.x * 256 + threadIdx.x;
    if (idx >= (long)BATCH * N) return;
    int m = (int)(idx / N), n = (int)(idx % N);
    int acc = bi[n];
    const int* ap = (const int*)(a5 + (size_t)m * 256);
    const int* wp = (const int*)(qw + (size_t)n * 256);
    for (int k = 0; k < 64; k++) {
        int aw = ap[k], ww = wp[k];
#if __has_builtin(__builtin_amdgcn_sdot4)
        acc = __builtin_amdgcn_sdot4(aw, ww, acc, false);
#else
        acc += (int)(int8_t)(aw) * (int)(int8_t)(ww)
             + (int)(int8_t)(aw >> 8)  * (int)(int8_t)(ww >> 8)
             + (int)(int8_t)(aw >> 16) * (int)(int8_t)(ww >> 16)
             + (int)(int8_t)(aw >> 24) * (int)(int8_t)(ww >> 24);
#endif
    }
    float y = (*sc_p) * (float)acc;
    if (mode == 0) {
        atomicMax(maxbits, __float_as_uint(fabsf(y)));
    } else {
        float q = fminf(7.f, fmaxf(-8.f, rintf(y * (*rq_p)))) * (*sq_p);
        out[(size_t)m * N + n] = q;
    }
}

extern "C" void kernel_launch(void* const* d_in, const int* in_sizes, int n_in,
                              void* d_out, int out_size, void* d_ws, size_t ws_size,
                              hipStream_t stream) {
    const float* x  = (const float*)d_in[0];
    const float* w1 = (const float*)d_in[1];  const float* b1 = (const float*)d_in[2];
    const float* w2 = (const float*)d_in[3];  const float* b2 = (const float*)d_in[4];
    const float* w3 = (const float*)d_in[5];  const float* b3 = (const float*)d_in[6];
    const float* w4 = (const float*)d_in[7];  const float* b4 = (const float*)d_in[8];
    const float* fw = (const float*)d_in[9];  const float* fb = (const float*)d_in[10];
    const float* bw = (const float*)d_in[11]; const float* bb = (const float*)d_in[12];
    const float* cw = (const float*)d_in[13]; const float* cb = (const float*)d_in[14];
    const float* lw = (const float*)d_in[15]; const float* lb = (const float*)d_in[16];
    float* O = (float*)d_out;

    char* ws = (char*)d_ws;
    float*    F = (float*)ws;
    int*      I = (int*)(ws + 4096);
    unsigned* U = (unsigned*)I;

    size_t off = 8192;
    auto alloc = [&](size_t bytes) -> char* {
        char* p = ws + off;
        off = (off + bytes + 255) & ~(size_t)255;
        return p;
    };
    int8_t* qwb = (int8_t*)alloc(4 * 256);
    int8_t* qwc = (int8_t*)alloc(2 * 256);
    int8_t* qwl = (int8_t*)alloc(10 * 256);
    int* wf1 = (int*)alloc((size_t)2  * 1  * 256 * 4);
    int* wf2 = (int*)alloc((size_t)4  * 5  * 256 * 4);
    int* wf3 = (int*)alloc((size_t)4  * 9  * 256 * 4);
    int* wf4 = (int*)alloc((size_t)8  * 4  * 256 * 4);
    int* wff = (int*)alloc((size_t)16 * 18 * 256 * 4);
    int8_t* qx  = (int8_t*)alloc((size_t)BATCH * 4 * 48 * 48);
    int8_t* a1  = (int8_t*)alloc((size_t)BATCH * 32 * 46 * 46);
    int8_t* a1p = (int8_t*)alloc((size_t)BATCH * 32 * 23 * 23);
    int8_t* a2  = (int8_t*)alloc((size_t)BATCH * 64 * 21 * 21);
    int8_t* a2p = (int8_t*)alloc((size_t)BATCH * 64 * 10 * 10);
    int8_t* a3  = (int8_t*)alloc((size_t)BATCH * 64 * 8 * 8);
    int8_t* a3p = (int8_t*)alloc((size_t)BATCH * 64 * 4 * 4);
    int8_t* a4  = (int8_t*)alloc((size_t)BATCH * 128 * 3 * 3);
    int8_t* a5  = (int8_t*)alloc((size_t)BATCH * 256);

    init_k<<<8, 256, 0, stream>>>((unsigned*)ws);

    long nx = (long)BATCH * 3 * 48 * 48;
    absmax_k<<<2048, 256, 0, stream>>>(x, nx, &U[I_ABSX]);
    pack_frag_k <<<1, 256, 0, stream>>>(w1, 32,  36,   4,   3, 3, 3, 0, &F[F_SW1], wf1);
    pack_frag_k <<<1, 256, 0, stream>>>(w2, 64,  288,  32,  32, 3, 3, 0, &F[F_SW2], wf2);
    pack_frag_k <<<1, 256, 0, stream>>>(w3, 64,  576,  64,  64, 3, 3, 0, &F[F_SW3], wf3);
    pack_frag_k <<<1, 256, 0, stream>>>(w4, 128, 256,  64,  64, 2, 2, 0, &F[F_SW4], wf4);
    pack_frag_k <<<1, 256, 0, stream>>>(fw, 256, 1152, 128, 128, 3, 3, 1, &F[F_SWF], wff);
    pack_plain_k<<<1, 256, 0, stream>>>(bw, 4 * 256,  &F[F_SWB], qwb);
    pack_plain_k<<<1, 256, 0, stream>>>(cw, 2 * 256,  &F[F_SWC], qwc);
    pack_plain_k<<<1, 256, 0, stream>>>(lw, 10 * 256, &F[F_SWL], qwl);

    scale_div_k<<<1, 1, 0, stream>>>(&U[I_ABSX], &F[F_S0], &F[F_R0], 7.f);
    long nq = (long)BATCH * 4 * 48 * 48;
    quantx_k<<<(int)((nq + 255) / 256), 256, 0, stream>>>(x, &F[F_R0], qx, nq);

    bias_prep_k<<<1, 256, 0, stream>>>(b1, &F[F_S0], &F[F_SW1], &F[F_SC1], I + BI1, 32);
    int M1 = BATCH * 46 * 46;
    dim3 g1(M1 / 64, 32 / 32);
    gemmq_k<true, false, 36, 32, 4, 2, 48, 48, 46, 46, 3, 3><<<g1, 128, 0, stream>>>(
        qx, wf1, I + BI1, &F[F_SC1], &I[I_MAX1], &F[F_RA1], a1, M1, 0);
    scale_act_k<<<1, 1, 0, stream>>>(&F[F_SC1], &I[I_MAX1], &F[F_SA1], &F[F_RA1]);
    gemmq_k<true, false, 36, 32, 4, 2, 48, 48, 46, 46, 3, 3><<<g1, 128, 0, stream>>>(
        qx, wf1, I + BI1, &F[F_SC1], &I[I_MAX1], &F[F_RA1], a1, M1, 1);
    long tp1 = (long)BATCH * 32 * 23 * 23;
    pool_k<<<(int)((tp1 + 255) / 256), 256, 0, stream>>>(a1, a1p, tp1, 32, 46, 46, 23, 23, 3, 2);

    bias_prep_k<<<1, 256, 0, stream>>>(b2, &F[F_SA1], &F[F_SW2], &F[F_SC2], I + BI2, 64);
    int M2 = BATCH * 21 * 21;
    dim3 g2(M2 / 64, 64 / 32);
    gemmq_k<false, false, 288, 64, 32, 5, 23, 23, 21, 21, 3, 3><<<g2, 128, 0, stream>>>(
        a1p, wf2, I + BI2, &F[F_SC2], &I[I_MAX2], &F[F_RA2], a2, M2, 0);
    scale_act_k<<<1, 1, 0, stream>>>(&F[F_SC2], &I[I_MAX2], &F[F_SA2], &F[F_RA2]);
    gemmq_k<false, false, 288, 64, 32, 5, 23, 23, 21, 21, 3, 3><<<g2, 128, 0, stream>>>(
        a1p, wf2, I + BI2, &F[F_SC2], &I[I_MAX2], &F[F_RA2], a2, M2, 1);
    long tp2 = (long)BATCH * 64 * 10 * 10;
    pool_k<<<(int)((tp2 + 255) / 256), 256, 0, stream>>>(a2, a2p, tp2, 64, 21, 21, 10, 10, 3, 2);

    bias_prep_k<<<1, 256, 0, stream>>>(b3, &F[F_SA2], &F[F_SW3], &F[F_SC3], I + BI3, 64);
    int M3 = BATCH * 8 * 8;
    dim3 g3(M3 / 64, 64 / 32);
    gemmq_k<false, false, 576, 64, 64, 6, 10, 10, 8, 8, 3, 3><<<g3, 128, 0, stream>>>(
        a2p, wf3, I + BI3, &F[F_SC3], &I[I_MAX3], &F[F_RA3], a3, M3, 0);
    scale_act_k<<<1, 1, 0, stream>>>(&F[F_SC3], &I[I_MAX3], &F[F_SA3], &F[F_RA3]);
    gemmq_k<false, false, 576, 64, 64, 6, 10, 10, 8, 8, 3, 3><<<g3, 128, 0, stream>>>(
        a2p, wf3, I + BI3, &F[F_SC3], &I[I_MAX3], &F[F_RA3], a3, M3, 1);
    long tp3 = (long)BATCH * 64 * 4 * 4;
    pool_k<<<(int)((tp3 + 255) / 256), 256, 0, stream>>>(a3, a3p, tp3, 64, 8, 8, 4, 4, 2, 2);

    bias_prep_k<<<1, 256, 0, stream>>>(b4, &F[F_SA3], &F[F_SW4], &F[F_SC4], I + BI4, 128);
    int M4 = BATCH * 3 * 3;
    dim3 g4(M4 / 64, 128 / 32);
    gemmq_k<false, false, 256, 128, 64, 6, 4, 4, 3, 3, 2, 2><<<g4, 128, 0, stream>>>(
        a3p, wf4, I + BI4, &F[F_SC4], &I[I_MAX4], &F[F_RA4], a4, M4, 0);
    scale_act_k<<<1, 1, 0, stream>>>(&F[F_SC4], &I[I_MAX4], &F[F_SA4], &F[F_RA4]);
    gemmq_k<false, false, 256, 128, 64, 6, 4, 4, 3, 3, 2, 2><<<g4, 128, 0, stream>>>(
        a3p, wf4, I + BI4, &F[F_SC4], &I[I_MAX4], &F[F_RA4], a4, M4, 1);

    bias_prep_k<<<1, 256, 0, stream>>>(fb, &F[F_SA4], &F[F_SWF], &F[F_SCF], I + BIF, 256);
    dim3 g5(BATCH / 64, 256 / 32);
    gemmq_k<false, true, 1152, 256, 128, 7, 1, 1, 1, 1, 1, 1><<<g5, 128, 0, stream>>>(
        a4, wff, I + BIF, &F[F_SCF], &I[I_MAX5], &F[F_RA5], a5, BATCH, 0);
    scale_act_k<<<1, 1, 0, stream>>>(&F[F_SCF], &I[I_MAX5], &F[F_SA5], &F[F_RA5]);
    gemmq_k<false, true, 1152, 256, 128, 7, 1, 1, 1, 1, 1, 1><<<g5, 128, 0, stream>>>(
        a4, wff, I + BIF, &F[F_SCF], &I[I_MAX5], &F[F_RA5], a5, BATCH, 1);

    bias_prep_k<<<1, 256, 0, stream>>>(bb, &F[F_SA5], &F[F_SWB], &F[F_SCB], I + BIB, 4);
    int hb = (BATCH * 4 + 255) / 256;
    head_k<<<hb, 256, 0, stream>>>(a5, qwb, I + BIB, &F[F_SCB], &U[I_ABSB], &F[F_SQB], &F[F_RQB], O, 4, 0);
    scale_div_k<<<1, 1, 0, stream>>>(&U[I_ABSB], &F[F_SQB], &F[F_RQB], 7.f);
    head_k<<<hb, 256, 0, stream>>>(a5, qwb, I + BIB, &F[F_SCB], &U[I_ABSB], &F[F_SQB], &F[F_RQB], O, 4, 1);

    bias_prep_k<<<1, 256, 0, stream>>>(cb, &F[F_SA5], &F[F_SWC], &F[F_SCC], I + BIC, 2);
    int hc = (BATCH * 2 + 255) / 256;
    head_k<<<hc, 256, 0, stream>>>(a5, qwc, I + BIC, &F[F_SCC], &U[I_ABSC], &F[F_SQC], &F[F_RQC], O + 16384, 2, 0);
    scale_div_k<<<1, 1, 0, stream>>>(&U[I_ABSC], &F[F_SQC], &F[F_RQC], 7.f);
    head_k<<<hc, 256, 0, stream>>>(a5, qwc, I + BIC, &F[F_SCC], &U[I_ABSC], &F[F_SQC], &F[F_RQC], O + 16384, 2, 1);

    bias_prep_k<<<1, 256, 0, stream>>>(lb, &F[F_SA5], &F[F_SWL], &F[F_SCL], I + BIL, 10);
    int hl = (BATCH * 10 + 255) / 256;
    head_k<<<hl, 256, 0, stream>>>(a5, qwl, I + BIL, &F[F_SCL], &U[I_ABSL], &F[F_SQL], &F[F_RQL], O + 24576, 10, 0);
    scale_div_k<<<1, 1, 0, stream>>>(&U[I_ABSL], &F[F_SQL], &F[F_RQL], 7.f);
    head_k<<<hl, 256, 0, stream>>>(a5, qwl, I + BIL, &F[F_SCL], &U[I_ABSL], &F[F_SQL], &F[F_RQL], O + 24576, 10, 1);
}
